// EoMT_70944269795988
// MI455X (gfx1250) — matrix operand order repaired and checked by reading, not compiled, not run on hardware
//
#include <hip/hip_runtime.h>
#include <hip/hip_bf16.h>

typedef __bf16 bf16;
typedef __attribute__((ext_vector_type(16))) __bf16 v16bf;
typedef __attribute__((ext_vector_type(8)))  float  v8f;
typedef unsigned __attribute__((ext_vector_type(4))) v4u;
typedef int      __attribute__((ext_vector_type(8))) v8i;
typedef int      __attribute__((ext_vector_type(4))) v4i;

#define NQ    200
#define PREF  5
#define PP    1600
#define NN    1805
#define DD    1024
#define NHD   16
#define DHD   64
#define NBLK  4
#define NCL   151
#define NEGB  (-1.0e9f)

union Frag { v16bf v; float4 f[2]; };

static __device__ inline v8f vzero8() {
  v8f z = {0.f,0.f,0.f,0.f,0.f,0.f,0.f,0.f};
  return z;
}

// ------------------------------------------------------- TDM 2D tile load ---
// Issue a TENSOR_LOAD_TO_LDS of a 2D bf16 tile:
//   rows   : tile_h rows of tile_w elements, row stride = stride_e elements
//   OOB    : rows >= tensor_h are zero-filled by the TDM (free edge handling)
//   padding: pad_amt_code+1 DWORDs inserted every 2^(pad_int_code+1) DWORDs,
//            reproducing the bank-conflict-avoiding LDS row pitch.
// Descriptor packing per CDNA5 ISA (groups 2/3 zero:
// 2D tensor). All inputs wave-uniform -> SGPR groups. Issue from ONE wave.
// This toolchain's builtin takes the 6-arg (clang-23) form:
//   (uint32x4 g0, int32x8 g1, int32x4, int32x4, int32x8, i32 cpol)
static __device__ inline void tdm_load_2d_bf16(
    unsigned lds_addr, const void* gaddr,
    unsigned tile_w, unsigned tile_h, unsigned tensor_h,
    unsigned stride_e, unsigned pad_int_code, unsigned pad_amt_code)
{
  unsigned long long ga = (unsigned long long)(size_t)gaddr;
  v4u g0;
  g0[0] = 1u;                                        // count=1, user mode
  g0[1] = lds_addr;                                  // LDS byte address
  g0[2] = (unsigned)(ga & 0xffffffffu);              // global_addr[31:0]
  g0[3] = (unsigned)((ga >> 32) & 0x1ffffffu)        // global_addr[56:32]
        | (2u << 30);                                // type = 2 ("image")
  v8i g1;
  g1[0] = (int)((1u << 16)                           // data_size = 2 bytes
        | (1u << 20)                                 // pad_enable
        | (pad_int_code << 22)                       // pad_interval code
        | (pad_amt_code << 25));                     // pad_amount code
  g1[1] = (int)((tile_w & 0xffffu) << 16);           // tensor_dim0[15:0]=tile_w
  g1[2] = (int)(((tile_w >> 16) & 0xffffu)           // tensor_dim0[31:16]
        | ((tensor_h & 0xffffu) << 16));             // tensor_dim1[15:0]
  g1[3] = (int)(((tensor_h >> 16) & 0xffffu)         // tensor_dim1[31:16]
        | ((tile_w & 0xffffu) << 16));               // tile_dim0
  g1[4] = (int)(tile_h & 0xffffu);                   // tile_dim1 (tile_dim2=0)
  g1[5] = (int)stride_e;                             // tensor_dim0_stride[31:0]
  g1[6] = 0;                                         // stride hi / dim1_stride
  g1[7] = 0;
  v4i z4 = {0, 0, 0, 0};
  v8i z8 = {0, 0, 0, 0, 0, 0, 0, 0};
  __builtin_amdgcn_tensor_load_to_lds(g0, g1, z4, z4, z8, 0);
}

static __device__ inline unsigned lds_off(const void* p) {
  // LDS aperture: generic address low 32 bits == LDS byte offset (ISA §10.2)
  return (unsigned)(size_t)p;
}

// ---------------------------------------------------------------- concat ----
__global__ void __launch_bounds__(256) concat_kernel(
    const float* __restrict__ xtok, const float* __restrict__ qw,
    float* __restrict__ X)
{
  int row = blockIdx.x, t = threadIdx.x;
  const float* src = (row < NQ) ? (qw + (size_t)row * DD)
                                : (xtok + (size_t)(row - NQ) * DD);
  float4 v = *(const float4*)(src + t * 4);
  *(float4*)(X + (size_t)row * DD + t * 4) = v;
}

// -------------------------------------------------------------- layernorm ---
__global__ void __launch_bounds__(256) ln_kernel(
    const float* __restrict__ X, const float* __restrict__ g,
    const float* __restrict__ b, bf16* __restrict__ out)
{
  int row = blockIdx.x, t = threadIdx.x;
  const float* x = X + (size_t)row * DD;
  float4 v = *(const float4*)(x + t * 4);
  float vv[4] = {v.x, v.y, v.z, v.w};
  float s = 0.f, ss = 0.f;
  for (int j = 0; j < 4; j++) { s += vv[j]; ss += vv[j] * vv[j]; }
  for (int o = 16; o >= 1; o >>= 1) {
    s  += __shfl_xor(s,  o, 32);
    ss += __shfl_xor(ss, o, 32);
  }
  __shared__ float ps[8], pq[8];
  if ((t & 31) == 0) { ps[t >> 5] = s; pq[t >> 5] = ss; }
  __syncthreads();
  s = 0.f; ss = 0.f;
  for (int i = 0; i < 8; i++) { s += ps[i]; ss += pq[i]; }
  float mean = s * (1.0f / DD);
  float var  = ss * (1.0f / DD) - mean * mean;
  float rs   = rsqrtf(var + 1e-6f);
  bf16* o = out + (size_t)row * DD;
  for (int j = 0; j < 4; j++) {
    int c = t * 4 + j;
    o[c] = (bf16)((vv[j] - mean) * rs * g[c] + b[c]);
  }
}

// -------------------------------------------- keep-bitmask (transposed) -----
__global__ void __launch_bounds__(256) keep_kernel(
    const float* __restrict__ mlog, unsigned* __restrict__ keepT)
{
  int idx = blockIdx.x * 256 + threadIdx.x;   // PP*8 = 12800 words
  if (idx >= PP * 8) return;
  int patch = idx >> 3, w = idx & 7;
  unsigned bits = 0u;
  for (int j = 0; j < 32; j++) {
    int q = w * 32 + j;
    bool keep = true;
    if (q < NQ) keep = mlog[(size_t)q * PP + patch] > 0.f;
    bits |= (keep ? 1u : 0u) << j;
  }
  keepT[idx] = bits;
}

// ------------------------------------------------------------- WMMA GEMM ----
// C[M,Nout] = epi(A_bf16[M,K] @ B + bias)
// A tiles staged by the Tensor Data Mover, double-buffered (wave 0 issues the
// k+1 tile while all waves compute on tile k; TENSORcnt<=1 keeps one in
// flight). Weights (fp32) staged manually with bf16 conversion + prefetch.
enum { EPI_F32 = 0, EPI_BF16 = 1, EPI_GELU_BF16 = 2, EPI_RESID = 3 };

#define BM  128
#define BN  128
#define BKK 32
#define LDA 40   // 32 + 4-DWORD TDM pad (pad_interval code 3 = 16 DWORDs)
#define LDB 40

template<int EPI, bool BT>
__global__ void __launch_bounds__(256) gemm_kernel(
    const bf16*  __restrict__ A,  const float* __restrict__ W,
    const bf16*  __restrict__ Bt, int M, int Nout, int K,
    const float* __restrict__ bias,
    float* __restrict__ Cf, bf16* __restrict__ Cb, int ldc,
    const float* __restrict__ ls, float* __restrict__ Xres)
{
  __shared__ bf16 As[2][BM * LDA];
  __shared__ bf16 Bs[BN * LDB];

  int t    = threadIdx.x;
  int n0   = blockIdx.x * BN;
  int m0   = blockIdx.y * BM;
  int wave = t >> 5, lane = t & 31;
  int lh   = lane & 15, hb = lane >> 4;
  int wm   = (wave >> 1) * 32;
  int wn   = (wave & 1) * 64;

  v8f acc[2][4];
  for (int mi = 0; mi < 2; mi++)
    for (int ni = 0; ni < 4; ni++) acc[mi][ni] = vzero8();

  const bf16* Abase = A + (size_t)m0 * K;
  unsigned mrem = (unsigned)(M - m0);           // TDM zero-fills rows >= mrem
  int nk = K / BKK;

  if (t < 32)                                   // prefetch A tile 0
    tdm_load_2d_bf16(lds_off(As[0]), Abase, BKK, BM, mrem, (unsigned)K, 3, 3);

  for (int kt = 0; kt < nk; kt++) {
    int k0 = kt * BKK;
    // ---- stage B tile as Bs[n][k] (manual: fp32 -> bf16 conversion) ----
    if (!BT) {
      if ((Nout & 127) == 0) {
        int kk = t >> 3;
        int nb = (t & 7) * 16;
        const float* wp = W + (size_t)(k0 + kk) * Nout + (n0 + nb);
        if (k0 + BKK < K)
          __builtin_prefetch((const void*)(wp + (size_t)BKK * Nout), 0, 1);
        for (int jj = 0; jj < 16; jj += 4) {
          float4 w4 = *(const float4*)(wp + jj);
          Bs[(nb + jj + 0) * LDB + kk] = (bf16)w4.x;
          Bs[(nb + jj + 1) * LDB + kk] = (bf16)w4.y;
          Bs[(nb + jj + 2) * LDB + kk] = (bf16)w4.z;
          Bs[(nb + jj + 3) * LDB + kk] = (bf16)w4.w;
        }
      } else {                            // ragged N (class head)
        for (int j = 0; j < 16; j++) {
          int idx = j * 256 + t;
          int kk = idx >> 7, n = idx & 127;
          float w = 0.f;
          if (n0 + n < Nout) w = W[(size_t)(k0 + kk) * Nout + n0 + n];
          Bs[n * LDB + kk] = (bf16)w;
        }
      }
    } else {
      int nr = t >> 1, kc = (t & 1) * 16;
      float4 z = make_float4(0.f,0.f,0.f,0.f), y0 = z, y1 = z;
      if (n0 + nr < Nout) {
        const bf16* bp = Bt + (size_t)(n0 + nr) * K + k0 + kc;
        y0 = *(const float4*)bp;
        y1 = *(const float4*)(bp + 8);
      }
      *(float4*)&Bs[nr * LDB + kc]     = y0;
      *(float4*)&Bs[nr * LDB + kc + 8] = y1;
    }
    // ---- issue TDM for next A tile; ensure current A tile landed ----
    if (t < 32) {
      if (kt + 1 < nk) {
        tdm_load_2d_bf16(lds_off(As[(kt + 1) & 1]),
                         Abase + (size_t)(k0 + BKK), BKK, BM, mrem,
                         (unsigned)K, 3, 3);
        __builtin_amdgcn_s_wait_tensorcnt(1);   // A(kt) done, A(kt+1) in flight
      } else {
        __builtin_amdgcn_s_wait_tensorcnt(0);
      }
    }
    __syncthreads();

    // ---- fragments + 8 WMMAs ----
    const bf16* Asb = As[kt & 1];
    Frag a[2], b[4];
    for (int mi = 0; mi < 2; mi++) {
      const bf16* ap = &Asb[(wm + mi * 16 + lh) * LDA + hb * 8];
      a[mi].f[0] = *(const float4*)ap;
      a[mi].f[1] = *(const float4*)(ap + 16);
    }
    for (int ni = 0; ni < 4; ni++) {
      const bf16* bp = &Bs[(wn + ni * 16 + lh) * LDB + hb * 8];
      b[ni].f[0] = *(const float4*)bp;
      b[ni].f[1] = *(const float4*)(bp + 16);
    }
    for (int mi = 0; mi < 2; mi++)
      for (int ni = 0; ni < 4; ni++)
        acc[mi][ni] = __builtin_amdgcn_wmma_f32_16x16x32_bf16(
            false, a[mi].v, false, b[ni].v, (short)0, acc[mi][ni],
            false, false);
    __syncthreads();
  }

  // ---- epilogue (C layout: row = r + 8*hb, col = lh) ----
  for (int mi = 0; mi < 2; mi++) {
    int row = m0 + wm + mi * 16 + hb * 8;
    for (int ni = 0; ni < 4; ni++) {
      int col = n0 + wn + ni * 16 + lh;
      if (col >= Nout) continue;
      for (int r = 0; r < 8; r++) {
        int rr = row + r;
        if (rr >= M) break;
        float v = acc[mi][ni][r];
        if (bias) v += bias[col];
        if (EPI == EPI_F32) {
          Cf[(size_t)rr * ldc + col] = v;
        } else if (EPI == EPI_BF16) {
          Cb[(size_t)rr * ldc + col] = (bf16)v;
        } else if (EPI == EPI_GELU_BF16) {
          float gv = 0.5f * v * (1.0f + erff(v * 0.70710678118654752f));
          Cb[(size_t)rr * ldc + col] = (bf16)gv;
        } else { // EPI_RESID: X += ls * (A@W + b)
          Xres[(size_t)rr * ldc + col] += ls[col] * v;
        }
      }
    }
  }
}

// -------------------------------------------------- flash attention ---------
// grid = (ceil(N/64), heads); 128 threads = 4 waves, 16 queries per wave.
// K tile staged by TDM (zero-filled past N), V transposed manually in its
// shadow; mask bias branch-free from the transposed keep bitmask.
#define SKS 72   // 64 + 4-DWORD TDM pad (pad_interval code 4 = 32 DWORDs)
#define SVS 40
__global__ void __launch_bounds__(128) attn_kernel(
    const bf16* __restrict__ qkv, const unsigned* __restrict__ keepT,
    bf16* __restrict__ Ob)
{
  __shared__ bf16 Ks[32 * SKS];        // Ks[key][dh]
  __shared__ bf16 Vt[64 * SVS];        // Vt[dh][key]
  __shared__ bf16 Pl[4][16 * SVS];     // per-wave P tile [q][key]

  int t    = threadIdx.x;
  int h    = blockIdx.y;
  int q0   = blockIdx.x * 64;
  int wave = t >> 5, lane = t & 31;
  int lh   = lane & 15, hb = lane >> 4;
  int qb   = q0 + wave * 16;
  int qword = qb >> 5;
  int qbit  = (qb & 31) + hb * 8;

  // Q fragments kept in registers
  Frag qf[2];
  {
    int qrow = qb + lh;
    if (qrow < NN) {
      const bf16* qp = qkv + (size_t)qrow * 3 * DD + h * DHD;
      for (int kc = 0; kc < 2; kc++) {
        const bf16* p = qp + kc * 32 + hb * 8;
        qf[kc].f[0] = *(const float4*)p;
        qf[kc].f[1] = *(const float4*)(p + 16);
      }
    } else {
      float4 z = make_float4(0.f,0.f,0.f,0.f);
      for (int kc = 0; kc < 2; kc++) { qf[kc].f[0] = z; qf[kc].f[1] = z; }
    }
  }

  float m_run[8], l_run[8];
  v8f o_acc[4];
  for (int r = 0; r < 8; r++) { m_run[r] = -1.0e30f; l_run[r] = 0.f; }
  for (int ni = 0; ni < 4; ni++) o_acc[ni] = vzero8();
  const float scale = 0.125f;   // 1/sqrt(64)

  for (int kt = 0; kt < NN; kt += 32) {
    __syncthreads();
    // TDM: K tile 32 keys x 64 dh, zero-filled for keys >= NN
    if (t < 32)
      tdm_load_2d_bf16(lds_off(Ks),
                       qkv + (size_t)kt * 3 * DD + DD + h * DHD,
                       DHD, 32, (unsigned)(NN - kt), 3 * DD, 4, 3);
    // manual transposed V staging overlaps the TDM flight
    {
      int key = t >> 2;
      int d0  = (t & 3) * 16;
      int gk  = kt + key;
      float4 z = make_float4(0.f,0.f,0.f,0.f);
      union { float4 f[2]; bf16 e[16]; } u;
      u.f[0] = z; u.f[1] = z;
      if (gk < NN) {
        const bf16* vp = qkv + (size_t)gk * 3 * DD + 2 * DD + h * DHD + d0;
        u.f[0] = *(const float4*)vp;
        u.f[1] = *(const float4*)(vp + 8);
      }
      for (int j = 0; j < 16; j++) Vt[(d0 + j) * SVS + key] = u.e[j];
    }
    if (t < 32) __builtin_amdgcn_s_wait_tensorcnt(0);
    __syncthreads();

    // S(16x32) = Q @ K^T
    v8f s[2];
    for (int nf = 0; nf < 2; nf++) {
      Frag bk[2];
      for (int kc = 0; kc < 2; kc++) {
        const bf16* bp = &Ks[(nf * 16 + lh) * SKS + kc * 32 + hb * 8];
        bk[kc].f[0] = *(const float4*)bp;
        bk[kc].f[1] = *(const float4*)(bp + 16);
      }
      v8f sv = vzero8();
      sv = __builtin_amdgcn_wmma_f32_16x16x32_bf16(
          false, qf[0].v, false, bk[0].v, (short)0, sv, false, false);
      sv = __builtin_amdgcn_wmma_f32_16x16x32_bf16(
          false, qf[1].v, false, bk[1].v, (short)0, sv, false, false);
      s[nf] = sv;
    }

    // scale + branch-free mask bias + key padding
    for (int nf = 0; nf < 2; nf++) {
      int key  = kt + nf * 16 + lh;
      int prow = key - (NQ + PREF);
      bool valid   = (key < NN);
      bool inpatch = valid && (prow >= 0);
      int cp = prow < 0 ? 0 : (prow >= PP ? PP - 1 : prow);
      unsigned kw   = keepT[cp * 8 + qword];
      unsigned bits = inpatch ? kw : 0xffffffffu;
      for (int r = 0; r < 8; r++) {
        float v = s[nf][r] * scale;
        v += ((bits >> (qbit + r)) & 1u) ? 0.f : NEGB;
        s[nf][r] = valid ? v : -1.0e30f;
      }
    }

    // online softmax (row reductions within each 16-lane half)
    for (int r = 0; r < 8; r++) {
      float mx = fmaxf(s[0][r], s[1][r]);
      for (int o = 8; o >= 1; o >>= 1) mx = fmaxf(mx, __shfl_xor(mx, o, 32));
      float m_new = fmaxf(m_run[r], mx);
      float corr  = __expf(m_run[r] - m_new);
      float p0 = __expf(s[0][r] - m_new);
      float p1 = __expf(s[1][r] - m_new);
      float rsum = p0 + p1;
      for (int o = 8; o >= 1; o >>= 1) rsum += __shfl_xor(rsum, o, 32);
      l_run[r] = l_run[r] * corr + rsum;
      m_run[r] = m_new;
      for (int ni = 0; ni < 4; ni++) o_acc[ni][r] *= corr;
      bf16* pl = &Pl[wave][(hb * 8 + r) * SVS];
      pl[lh]      = (bf16)p0;
      pl[16 + lh] = (bf16)p1;
    }

    // O(16x64) += P(16x32) @ V(32x64)
    {
      Frag pa;
      const bf16* pp = &Pl[wave][lh * SVS + hb * 8];
      pa.f[0] = *(const float4*)pp;
      pa.f[1] = *(const float4*)(pp + 16);
      for (int ni = 0; ni < 4; ni++) {
        Frag vb;
        const bf16* vp = &Vt[(ni * 16 + lh) * SVS + hb * 8];
        vb.f[0] = *(const float4*)vp;
        vb.f[1] = *(const float4*)(vp + 16);
        o_acc[ni] = __builtin_amdgcn_wmma_f32_16x16x32_bf16(
            false, pa.v, false, vb.v, (short)0, o_acc[ni], false, false);
      }
    }
  }

  // normalize + store O (bf16, feeds proj GEMM)
  for (int r = 0; r < 8; r++) {
    int qrow = qb + hb * 8 + r;
    if (qrow >= NN) continue;
    float inv = 1.0f / l_run[r];
    for (int ni = 0; ni < 4; ni++) {
      int col = h * DHD + ni * 16 + lh;
      Ob[(size_t)qrow * DD + col] = (bf16)(o_acc[ni][r] * inv);
    }
  }
}

// ------------------------------------------------------------------ host ----
extern "C" void kernel_launch(void* const* d_in, const int* in_sizes, int n_in,
                              void* d_out, int out_size, void* d_ws,
                              size_t ws_size, hipStream_t stream)
{
  (void)in_sizes; (void)n_in; (void)out_size; (void)ws_size;
  const float* x_tokens = (const float*)d_in[0];
  const float* q_weight = (const float*)d_in[1];
  const float* norm_g   = (const float*)d_in[2];
  const float* norm_b   = (const float*)d_in[3];
  const float* class_W  = (const float*)d_in[4];
  const float* class_b  = (const float*)d_in[5];
  const float* mask_W1  = (const float*)d_in[6];
  const float* mask_b1  = (const float*)d_in[7];
  const float* mask_W2  = (const float*)d_in[8];
  const float* mask_b2  = (const float*)d_in[9];
  const float* mask_W3  = (const float*)d_in[10];
  const float* mask_b3  = (const float*)d_in[11];
  const float* ln1_g    = (const float*)d_in[12];
  const float* ln1_b    = (const float*)d_in[13];
  const float* ln2_g    = (const float*)d_in[14];
  const float* ln2_b    = (const float*)d_in[15];
  const float* qkv_W    = (const float*)d_in[16];
  const float* qkv_b    = (const float*)d_in[17];
  const float* proj_W   = (const float*)d_in[18];
  const float* proj_b   = (const float*)d_in[19];
  const float* ls1      = (const float*)d_in[20];
  const float* ls2      = (const float*)d_in[21];
  const float* fc1_W    = (const float*)d_in[22];
  const float* fc1_b    = (const float*)d_in[23];
  const float* fc2_W    = (const float*)d_in[24];
  const float* fc2_b    = (const float*)d_in[25];

  float* outp = (float*)d_out;
  const size_t MASKS_TOT = (size_t)(NBLK + 1) * NQ * PP;

  // workspace carve (~42 MB, every byte written before read)
  char* p = (char*)d_ws;
  auto carve = [&](size_t bytes) -> char* {
    char* r = p;
    p += (bytes + 255) & ~(size_t)255;
    return r;
  };
  float*    X     = (float*)   carve((size_t)NN * DD * 4);
  bf16*     NXb   = (bf16*)    carve((size_t)NN * DD * 2);
  bf16*     QKVb  = (bf16*)    carve((size_t)NN * 3 * DD * 2);
  bf16*     Obuf  = (bf16*)    carve((size_t)NN * DD * 2);
  bf16*     G1b   = (bf16*)    carve((size_t)NN * 4 * DD * 2);
  bf16*     f1b   = (bf16*)    carve((size_t)NQ * DD * 2);
  bf16*     f2b   = (bf16*)    carve((size_t)NQ * DD * 2);
  bf16*     f3b   = (bf16*)    carve((size_t)NQ * DD * 2);
  unsigned* keepT = (unsigned*)carve((size_t)PP * 8 * 4);

  dim3 blk(256);
  auto gg = [](int M, int Nn) { return dim3((Nn + BN - 1) / BN,
                                            (M + BM - 1) / BM); };

  concat_kernel<<<NN, 256, 0, stream>>>(x_tokens, q_weight, X);

  for (int i = 0; i <= NBLK; i++) {
    float* mask_out = outp + (size_t)i * NQ * PP;
    float* cls_out  = outp + MASKS_TOT + (size_t)i * NQ * NCL;

    // ---- predict head on LN(norm) of current stream ----
    ln_kernel<<<NN, 256, 0, stream>>>(X, norm_g, norm_b, NXb);
    gemm_kernel<EPI_GELU_BF16,false><<<gg(NQ, DD), blk, 0, stream>>>(
        NXb, mask_W1, nullptr, NQ, DD, DD, mask_b1,
        nullptr, f1b, DD, nullptr, nullptr);
    gemm_kernel<EPI_GELU_BF16,false><<<gg(NQ, DD), blk, 0, stream>>>(
        f1b, mask_W2, nullptr, NQ, DD, DD, mask_b2,
        nullptr, f2b, DD, nullptr, nullptr);
    gemm_kernel<EPI_BF16,false><<<gg(NQ, DD), blk, 0, stream>>>(
        f2b, mask_W3, nullptr, NQ, DD, DD, mask_b3,
        nullptr, f3b, DD, nullptr, nullptr);
    gemm_kernel<EPI_F32,true><<<gg(NQ, PP), blk, 0, stream>>>(
        f3b, nullptr, NXb + (size_t)(NQ + PREF) * DD, NQ, PP, DD, nullptr,
        mask_out, nullptr, PP, nullptr, nullptr);
    gemm_kernel<EPI_F32,false><<<gg(NQ, NCL), blk, 0, stream>>>(
        NXb, class_W, nullptr, NQ, NCL, DD, class_b,
        cls_out, nullptr, NCL, nullptr, nullptr);

    if (i == NBLK) break;

    // ---- transformer block i ----
    keep_kernel<<<(PP * 8 + 255) / 256, 256, 0, stream>>>(mask_out, keepT);
    ln_kernel<<<NN, 256, 0, stream>>>(
        X, ln1_g + (size_t)i * DD, ln1_b + (size_t)i * DD, NXb);
    gemm_kernel<EPI_BF16,false><<<gg(NN, 3 * DD), blk, 0, stream>>>(
        NXb, qkv_W + (size_t)i * DD * 3 * DD, nullptr, NN, 3 * DD, DD,
        qkv_b + (size_t)i * 3 * DD, nullptr, QKVb, 3 * DD, nullptr, nullptr);
    attn_kernel<<<dim3((NN + 63) / 64, NHD), dim3(128), 0, stream>>>(
        QKVb, keepT, Obuf);
    gemm_kernel<EPI_RESID,false><<<gg(NN, DD), blk, 0, stream>>>(
        Obuf, proj_W + (size_t)i * DD * DD, nullptr, NN, DD, DD,
        proj_b + (size_t)i * DD, nullptr, nullptr, DD,
        ls1 + (size_t)i * DD, X);
    ln_kernel<<<NN, 256, 0, stream>>>(
        X, ln2_g + (size_t)i * DD, ln2_b + (size_t)i * DD, NXb);
    gemm_kernel<EPI_GELU_BF16,false><<<gg(NN, 4 * DD), blk, 0, stream>>>(
        NXb, fc1_W + (size_t)i * DD * 4 * DD, nullptr, NN, 4 * DD, DD,
        fc1_b + (size_t)i * 4 * DD, nullptr, G1b, 4 * DD, nullptr, nullptr);
    gemm_kernel<EPI_RESID,false><<<gg(NN, DD), blk, 0, stream>>>(
        G1b, fc2_W + (size_t)i * 4 * DD * DD, nullptr, NN, DD, 4 * DD,
        fc2_b + (size_t)i * DD, nullptr, nullptr, DD,
        ls2 + (size_t)i * DD, X);
  }
}
